// HeteroGCNCiteerCrazy_40759239639279
// MI455X (gfx1250) — matrix operand order repaired and checked by reading, not compiled, not run on hardware
//
#include <hip/hip_runtime.h>
#include <hip/hip_bf16.h>

typedef __bf16 bf16;
typedef __attribute__((ext_vector_type(16))) bf16  v16bf;
typedef __attribute__((ext_vector_type(8)))  bf16  bf16x8;
typedef __attribute__((ext_vector_type(8)))  float v8f;
typedef __attribute__((ext_vector_type(2)))  float f32x2;

#define BM 64
#define BN 64
#define BK 32
#define LPAD 8  // bf16 elems of row padding (16B) to spread LDS banks

// ---------------------------------------------------------------------------
// Tiled bf16 WMMA GEMM: C[M,N] = A[M,K] @ B[K,N] (+bias[N]) (+C if accum)
// A,B,C f32 row-major. f32 accumulate via v_wmma_f32_16x16x32_bf16.
// K must be even (true for 602/300/128 here) so float2 loads stay aligned.
// ---------------------------------------------------------------------------
__global__ __launch_bounds__(256)
void k_gemm_bf16(const float* __restrict__ A, const float* __restrict__ B,
                 const float* __restrict__ bias, float* __restrict__ C,
                 int M, int K, int N, int accum)
{
    __shared__ __align__(16) bf16 As [BM][BK + LPAD];   // 64 x 32 (A tile)
    __shared__ __align__(16) bf16 BsT[BN][BK + LPAD];   // 64 x 32 (B tile, transposed)

    const int tid  = threadIdx.x;
    const int lane = tid & 31;
    const int wave = tid >> 5;
    const int wr   = wave & 3;      // wave row tile 0..3  (16 rows each)
    const int wc   = wave >> 2;     // wave col half 0..1  (32 cols each)
    const int h    = lane >> 4;     // lane half
    const int ln   = lane & 15;

    const int blockRow = blockIdx.x * BM;
    const int blockCol = blockIdx.y * BN;

    // staging thread mappings (fixed across k-steps)
    const int a_row = tid >> 2, a_seg = tid & 3;          // A: (row, 8-wide k segment)
    const int b_col = tid & 63, b_seg = tid >> 6;         // B: (col, 8-wide k segment)
    const int a_rg  = blockRow + a_row;
    const int b_cg  = blockCol + b_col;
    const bool a_row_ok = (a_rg < M);
    const size_t a_rowoff = (size_t)(a_row_ok ? a_rg : 0) * K;  // always-valid row

    v8f acc0 = {}; v8f acc1 = {};

    for (int k0 = 0; k0 < K; k0 += BK) {
        // --- stage A tile (f32 -> bf16) -----------------------------------
        {
            const int kb = k0 + a_seg * 8;
            bf16x8 outv;
            if (a_row_ok && (kb + 8 <= K)) {
                // fast path: 4x float2 (global_load_b64), 8B alignment guaranteed
                const f32x2* ap = (const f32x2*)(A + a_rowoff + kb);
                #pragma unroll
                for (int i = 0; i < 4; ++i) {
                    const f32x2 p = ap[i];
                    outv[2 * i + 0] = (bf16)p.x;
                    outv[2 * i + 1] = (bf16)p.y;
                }
            } else {
                // boundary: branchless clamped loads + select-zero
                #pragma unroll
                for (int i = 0; i < 8; ++i) {
                    const int kk = kb + i;
                    const int kc = kk < K ? kk : K - 1;         // clamped, always legal
                    float v = A[a_rowoff + kc];
                    v = (a_row_ok && kk < K) ? v : 0.0f;        // v_cndmask
                    outv[i] = (bf16)v;
                }
            }
            *(bf16x8*)&As[a_row][a_seg * 8] = outv;             // single ds_store_b128
        }
        // --- stage B tile transposed: column gather, coalesced loads ------
        {
            const int kb = k0 + b_seg * 8;
            bf16x8 outv;
            if (kb + 8 <= K) {
                const float* bp = B + (size_t)kb * N + b_cg;
                #pragma unroll
                for (int i = 0; i < 8; ++i) outv[i] = (bf16)bp[(size_t)i * N];
            } else {
                #pragma unroll
                for (int i = 0; i < 8; ++i) {
                    const int kk = kb + i;
                    const int kc = kk < K ? kk : K - 1;
                    float v = B[(size_t)kc * N + b_cg];
                    v = (kk < K) ? v : 0.0f;
                    outv[i] = (bf16)v;
                }
            }
            *(bf16x8*)&BsT[b_col][b_seg * 8] = outv;            // single ds_store_b128
        }
        __syncthreads();

        // --- A fragment: lanes 0-15 row M, K = 8h+e (e<8) / 16+8h+(e-8)
        v16bf afrag;
        {
            const int row = wr * 16 + ln;
            bf16x8 lo = *(const bf16x8*)&As[row][h * 8];
            bf16x8 hi = *(const bf16x8*)&As[row][16 + h * 8];
            #pragma unroll
            for (int i = 0; i < 8; ++i) { afrag[i] = lo[i]; afrag[8 + i] = hi[i]; }
        }
        // --- B fragments (two 16-col tiles): lane col N, K = 8h+e (e<8) / 16+8h+(e-8)
        #pragma unroll
        for (int nf = 0; nf < 2; ++nf) {
            const int col = wc * 32 + nf * 16 + ln;
            bf16x8 b0 = *(const bf16x8*)&BsT[col][h * 8];
            bf16x8 b1 = *(const bf16x8*)&BsT[col][16 + h * 8];
            v16bf bfrag;
            #pragma unroll
            for (int i = 0; i < 8; ++i) { bfrag[i] = b0[i]; bfrag[8 + i] = b1[i]; }
            if (nf == 0)
                acc0 = __builtin_amdgcn_wmma_f32_16x16x32_bf16(false, afrag, false, bfrag,
                                                               (short)0, acc0, false, false);
            else
                acc1 = __builtin_amdgcn_wmma_f32_16x16x32_bf16(false, afrag, false, bfrag,
                                                               (short)0, acc1, false, false);
        }
        __syncthreads();
    }

    // --- epilogue: C rows = h*8+i, cols = lane&15
    #pragma unroll
    for (int nf = 0; nf < 2; ++nf) {
        const v8f acc = nf ? acc1 : acc0;
        const int col = blockCol + wc * 32 + nf * 16 + ln;
        const float bv = bias ? bias[col] : 0.0f;
        #pragma unroll
        for (int i = 0; i < 8; ++i) {
            const int row = blockRow + wr * 16 + h * 8 + i;
            if (row < M) {
                float v = acc[i] + bv;
                const size_t idx = (size_t)row * N + col;
                if (accum) v += C[idx];
                C[idx] = v;
            }
        }
    }
}

// ---------------------------------------------------------------------------
// Elementwise / scatter helpers
// ---------------------------------------------------------------------------
__global__ void k_zero(float* __restrict__ p, long long n) {
    for (long long i = (long long)blockIdx.x * blockDim.x + threadIdx.x; i < n;
         i += (long long)gridDim.x * blockDim.x) p[i] = 0.0f;
}

__global__ void k_bias_init(float* __restrict__ out, long long rows,
                            const float* __restrict__ b1, const float* __restrict__ b2) {
    const long long total = rows * 128;
    for (long long i = (long long)blockIdx.x * blockDim.x + threadIdx.x; i < total;
         i += (long long)gridDim.x * blockDim.x) {
        const int f = (int)(i & 127);
        out[i] = b1[f] + (b2 ? b2[f] : 0.0f);
    }
}

// agg[dst] += x[src] over edges, F=128
__global__ void k_scatter_sum(const float* __restrict__ x, const int* __restrict__ src,
                              const int* __restrict__ dst, long long nE, float* __restrict__ out) {
    const long long total = nE * 128;
    for (long long i = (long long)blockIdx.x * blockDim.x + threadIdx.x; i < total;
         i += (long long)gridDim.x * blockDim.x) {
        const long long e = i >> 7;
        const int f = (int)(i & 127);
        const int s = src[e], d = dst[e];
        atomicAdd(&out[(size_t)d * 128 + f], x[(size_t)s * 128 + f]);
    }
}

// y = LN(relu(h)) * g + b + resid   (in place, one wave32 per 128-wide row)
__global__ __launch_bounds__(256)
void k_ln_relu_res(float* __restrict__ h, const float* __restrict__ resid,
                   const float* __restrict__ g, const float* __restrict__ b, int rows) {
    const int wave = threadIdx.x >> 5, lane = threadIdx.x & 31;
    const int row = blockIdx.x * 8 + wave;
    if (row >= rows) return;
    const size_t off = (size_t)row * 128 + lane * 4;
    float4 v = *(const float4*)(h + off);
    const float r0 = fmaxf(v.x, 0.f), r1 = fmaxf(v.y, 0.f),
                r2 = fmaxf(v.z, 0.f), r3 = fmaxf(v.w, 0.f);
    float s = r0 + r1 + r2 + r3;
    #pragma unroll
    for (int o = 16; o > 0; o >>= 1) s += __shfl_xor(s, o, 32);
    const float mu = s * (1.0f / 128.0f);
    const float d0 = r0 - mu, d1 = r1 - mu, d2 = r2 - mu, d3 = r3 - mu;
    float q = d0 * d0 + d1 * d1 + d2 * d2 + d3 * d3;
    #pragma unroll
    for (int o = 16; o > 0; o >>= 1) q += __shfl_xor(q, o, 32);
    const float inv = rsqrtf(q * (1.0f / 128.0f) + 1e-5f);
    const int f = lane * 4;
    const float4 gg = *(const float4*)(g + f);
    const float4 bb = *(const float4*)(b + f);
    const float4 rr = *(const float4*)(resid + off);
    float4 o4;
    o4.x = d0 * inv * gg.x + bb.x + rr.x;
    o4.y = d1 * inv * gg.y + bb.y + rr.y;
    o4.z = d2 * inv * gg.z + bb.z + rr.z;
    o4.w = d3 * inv * gg.w + bb.w + rr.w;
    *(float4*)(h + off) = o4;
}

// was[h,c] = sum_f W[c, h*128+f] * a_s[h,f] ; wad likewise (one 256-thread block)
__global__ void k_attn_vec(const float* __restrict__ W, const float* __restrict__ as_,
                           const float* __restrict__ ad_, float* __restrict__ was,
                           float* __restrict__ wad) {
    const int t = threadIdx.x, hh = t >> 7, c = t & 127;
    float s1 = 0.f, s2 = 0.f;
    for (int f = 0; f < 128; ++f) {
        const float w = W[(size_t)c * 256 + hh * 128 + f];
        s1 += w * as_[hh * 128 + f];
        s2 += w * ad_[hh * 128 + f];
    }
    was[hh * 128 + c] = s1;
    wad[hh * 128 + c] = s2;
}

// out[n,h] = x[n,:] . wa[h,:]
__global__ void k_attn_score(const float* __restrict__ x, const float* __restrict__ wa,
                             float* __restrict__ out, long long Nn) {
    const long long total = Nn * 2;
    for (long long i = (long long)blockIdx.x * blockDim.x + threadIdx.x; i < total;
         i += (long long)gridDim.x * blockDim.x) {
        const long long n = i >> 1; const int hh = (int)(i & 1);
        const float* xr = x + (size_t)n * 128;
        const float* wr = wa + hh * 128;
        float s = 0.f;
        #pragma unroll 4
        for (int c = 0; c < 128; ++c) s += xr[c] * wr[c];
        out[n * 2 + hh] = s;
    }
}

__global__ void k_init_mz(float* __restrict__ m, float* __restrict__ z, long long n) {
    for (long long i = (long long)blockIdx.x * blockDim.x + threadIdx.x; i < n;
         i += (long long)gridDim.x * blockDim.x) { m[i] = -1e30f; z[i] = 0.0f; }
}

__device__ inline void atomicMaxF(float* addr, float val) {
    if (val >= 0.0f) atomicMax((int*)addr, __float_as_int(val));
    else             atomicMin((unsigned int*)addr, (unsigned int)__float_as_int(val));
}

// e = leaky_relu(es[src]+ed[dst]); store raw logit; m[dst] = max(m, e)
__global__ void k_edge_max(const float* __restrict__ es, const float* __restrict__ ed,
                           const int* __restrict__ src, const int* __restrict__ dst,
                           long long nE, float* __restrict__ ew, float* __restrict__ m) {
    const long long total = nE * 2;
    for (long long i = (long long)blockIdx.x * blockDim.x + threadIdx.x; i < total;
         i += (long long)gridDim.x * blockDim.x) {
        const long long e = i >> 1; const int hh = (int)(i & 1);
        const int s = src[e], d = dst[e];
        float x = es[(size_t)s * 2 + hh] + ed[(size_t)d * 2 + hh];
        x = (x > 0.0f) ? x : 0.2f * x;
        ew[i] = x;
        atomicMaxF(&m[(size_t)d * 2 + hh], x);
    }
}

// w = exp(e - m[dst]); z[dst] += w
__global__ void k_edge_exp(float* __restrict__ ew, const float* __restrict__ m,
                           const int* __restrict__ dst, long long nE, float* __restrict__ z) {
    const long long total = nE * 2;
    for (long long i = (long long)blockIdx.x * blockDim.x + threadIdx.x; i < total;
         i += (long long)gridDim.x * blockDim.x) {
        const long long e = i >> 1; const int hh = (int)(i & 1);
        const int d = dst[e];
        const float w = __expf(ew[i] - m[(size_t)d * 2 + hh]);
        ew[i] = w;
        atomicAdd(&z[(size_t)d * 2 + hh], w);
    }
}

// out[dst,f] += mean_h( alpha[e,h] * hs[src,h,f] )
__global__ void k_edge_scatter(const float* __restrict__ ew, const float* __restrict__ z,
                               const float* __restrict__ hs, const int* __restrict__ src,
                               const int* __restrict__ dst, long long nE,
                               float* __restrict__ out) {
    const long long total = nE * 128;
    for (long long i = (long long)blockIdx.x * blockDim.x + threadIdx.x; i < total;
         i += (long long)gridDim.x * blockDim.x) {
        const long long e = i >> 7;
        const int f = (int)(i & 127);
        const int s = src[e], d = dst[e];
        const float a0 = ew[e * 2 + 0] / (z[(size_t)d * 2 + 0] + 1e-16f);
        const float a1 = ew[e * 2 + 1] / (z[(size_t)d * 2 + 1] + 1e-16f);
        const float val = 0.5f * (a0 * hs[(size_t)s * 256 + f] +
                                  a1 * hs[(size_t)s * 256 + 128 + f]);
        atomicAdd(&out[(size_t)d * 128 + f], val);
    }
}

// ---------------------------------------------------------------------------
// Host orchestration
// ---------------------------------------------------------------------------
static inline unsigned gs_blocks(long long total, int bs) {
    long long b = (total + bs - 1) / bs;
    if (b > 1048576LL) b = 1048576LL;
    if (b < 1) b = 1;
    return (unsigned)b;
}

extern "C" void kernel_launch(void* const* d_in, const int* in_sizes, int n_in,
                              void* d_out, int out_size, void* d_ws, size_t ws_size,
                              hipStream_t stream) {
    (void)n_in; (void)out_size; (void)ws_size;
    // ---- inputs
    const float* x_paper   = (const float*)d_in[0];
    const float* x_author  = (const float*)d_in[1];
    const int* cites_src   = (const int*)d_in[2];
    const int* cites_dst   = (const int*)d_in[3];
    const int* writes_src  = (const int*)d_in[4];
    const int* writes_dst  = (const int*)d_in[5];
    const int* written_src = (const int*)d_in[6];
    const int* written_dst = (const int*)d_in[7];

    const int DP = in_sizes[8]  / 128;           // 602
    const int DA = in_sizes[10] / 128;           // 300
    const int NP = in_sizes[0] / DP;             // 100000
    const int NA = in_sizes[1] / DA;             // 50000
    const long long EC  = in_sizes[2];
    const long long EW  = in_sizes[4];
    const long long EWR = in_sizes[6];

    const float* in_W_paper  = (const float*)d_in[8];
    const float* in_b_paper  = (const float*)d_in[9];
    const float* in_W_author = (const float*)d_in[10];
    const float* in_b_author = (const float*)d_in[11];
    // per-relation parameter blocks: cites=12, writes=19, written=26
    auto P = [&](int i) { return (const float*)d_in[i]; };
    const int RC = 12, RW = 19, RR = 26;
    // node-type params: paper=33..38, author=39..44
    const float* n1_g_p = P(33); const float* n1_b_p = P(34);
    const float* n2_g_p = P(35); const float* n2_b_p = P(36);
    const float* out_W_p = P(37); const float* out_b_p = P(38);
    const float* n1_g_a = P(39); const float* n1_b_a = P(40);
    const float* n2_g_a = P(41); const float* n2_b_a = P(42);
    const float* out_W_a = P(43); const float* out_b_a = P(44);

    // ---- workspace arena (f32)
    float* W = (float*)d_ws;
    size_t o = 0;
    float* xp   = W + o; o += (size_t)NP * 128;
    float* xa   = W + o; o += (size_t)NA * 128;
    float* h1p  = W + o; o += (size_t)NP * 128;
    float* h1a  = W + o; o += (size_t)NA * 128;
    float* h2p  = W + o; o += (size_t)NP * 128;
    float* h2a  = W + o; o += (size_t)NA * 128;
    float* agA  = W + o; o += (size_t)NP * 128;
    float* agB  = W + o; o += (size_t)NP * 128;
    float* HS   = W + o; o += (size_t)NP * 256;
    float* ES   = W + o; o += (size_t)NP * 2;
    float* ED   = W + o; o += (size_t)NP * 2;
    float* Mb   = W + o; o += (size_t)NP * 2;
    float* Zb   = W + o; o += (size_t)NP * 2;
    long long Emax = EC > EW ? EC : EW; if (EWR > Emax) Emax = EWR;
    float* EB   = W + o; o += (size_t)Emax * 2;
    float* WAS  = W + o; o += 256;
    float* WAD  = W + o; o += 256;

    auto gemm = [&](const float* A, const float* B, const float* bias, float* C,
                    int M, int K, int N, int accum) {
        dim3 grid((M + BM - 1) / BM, N / BN);
        k_gemm_bf16<<<grid, 256, 0, stream>>>(A, B, bias, C, M, K, N, accum);
    };

    // ================= phase 0: input projections =================
    gemm(x_paper,  in_W_paper,  in_b_paper,  xp, NP, DP, 128, 0);
    gemm(x_author, in_W_author, in_b_author, xa, NA, DA, 128, 0);

    // ================= phase 1: GraphConv =================
    k_zero<<<gs_blocks((long long)NP * 128, 256), 256, 0, stream>>>(agA, (long long)NP * 128);
    k_zero<<<gs_blocks((long long)NP * 128, 256), 256, 0, stream>>>(agB, (long long)NP * 128);
    k_scatter_sum<<<gs_blocks(EC * 128, 256), 256, 0, stream>>>(xp, cites_src, cites_dst, EC, agA);
    k_scatter_sum<<<gs_blocks(EW * 128, 256), 256, 0, stream>>>(xa, writes_src, writes_dst, EW, agB);
    gemm(agA, P(RC + 0), P(RC + 2), h1p, NP, 128, 128, 0);   // agg @ Wn_cites + b
    gemm(xp,  P(RC + 1), nullptr,   h1p, NP, 128, 128, 1);   // += xp @ Wr_cites
    gemm(agB, P(RW + 0), P(RW + 2), h1p, NP, 128, 128, 1);   // += agg @ Wn_writes + b
    gemm(xp,  P(RW + 1), nullptr,   h1p, NP, 128, 128, 1);   // += xp @ Wr_writes

    k_zero<<<gs_blocks((long long)NA * 128, 256), 256, 0, stream>>>(agA, (long long)NA * 128);
    k_scatter_sum<<<gs_blocks(EWR * 128, 256), 256, 0, stream>>>(xp, written_src, written_dst, EWR, agA);
    gemm(agA, P(RR + 0), P(RR + 2), h1a, NA, 128, 128, 0);
    gemm(xa,  P(RR + 1), nullptr,   h1a, NA, 128, 128, 1);

    k_ln_relu_res<<<(NP + 7) / 8, 256, 0, stream>>>(h1p, xp, n1_g_p, n1_b_p, NP);
    k_ln_relu_res<<<(NA + 7) / 8, 256, 0, stream>>>(h1a, xa, n1_g_a, n1_b_a, NA);

    // ================= phase 2: GATConv =================
    k_bias_init<<<gs_blocks((long long)NP * 128, 256), 256, 0, stream>>>(h2p, NP, P(RC + 6), P(RW + 6));
    k_bias_init<<<gs_blocks((long long)NA * 128, 256), 256, 0, stream>>>(h2a, NA, P(RR + 6), nullptr);

    auto run_gat = [&](const float* xs, int Ns, const float* xd, int Nd,
                       const int* src, const int* dst, long long E,
                       const float* Wg, const float* a_s, const float* a_d, float* outAcc) {
        gemm(xs, Wg, nullptr, HS, Ns, 128, 256, 0);                         // hs = xs @ W
        k_attn_vec<<<1, 256, 0, stream>>>(Wg, a_s, a_d, WAS, WAD);          // fold a into W
        k_attn_score<<<gs_blocks((long long)Ns * 2, 256), 256, 0, stream>>>(xs, WAS, ES, Ns);
        k_attn_score<<<gs_blocks((long long)Nd * 2, 256), 256, 0, stream>>>(xd, WAD, ED, Nd);
        k_init_mz<<<gs_blocks((long long)Nd * 2, 256), 256, 0, stream>>>(Mb, Zb, (long long)Nd * 2);
        k_edge_max<<<gs_blocks(E * 2, 256), 256, 0, stream>>>(ES, ED, src, dst, E, EB, Mb);
        k_edge_exp<<<gs_blocks(E * 2, 256), 256, 0, stream>>>(EB, Mb, dst, E, Zb);
        k_edge_scatter<<<gs_blocks(E * 128, 256), 256, 0, stream>>>(EB, Zb, HS, src, dst, E, outAcc);
    };

    run_gat(h1p, NP, h1p, NP, cites_src,   cites_dst,   EC,  P(RC + 3), P(RC + 4), P(RC + 5), h2p);
    run_gat(h1a, NA, h1p, NP, writes_src,  writes_dst,  EW,  P(RW + 3), P(RW + 4), P(RW + 5), h2p);
    run_gat(h1p, NP, h1a, NA, written_src, written_dst, EWR, P(RR + 3), P(RR + 4), P(RR + 5), h2a);

    k_ln_relu_res<<<(NP + 7) / 8, 256, 0, stream>>>(h2p, h1p, n2_g_p, n2_b_p, NP);
    k_ln_relu_res<<<(NA + 7) / 8, 256, 0, stream>>>(h2a, h1a, n2_g_a, n2_b_a, NA);

    // ================= phase 3: output projections =================
    float* out_p = (float*)d_out;
    float* out_a = out_p + (size_t)NP * 64;
    gemm(h2p, out_W_p, out_b_p, out_p, NP, 128, 64, 0);
    gemm(h2a, out_W_a, out_b_a, out_a, NA, 128, 64, 0);
}
